// unit_gcn_26027501814614
// MI455X (gfx1250) — compile-verified, matrix operand order checked by reading, not run on hardware
//
#include <hip/hip_runtime.h>
#include <math.h>

// Shapes
#define NB   128
#define CC   64
#define TT   128
#define VV   25
#define HH   8
#define HID  64
#define P2   625            // V*V
#define KPAD 256            // 8 c-groups * 32 padded V
#define VPAD 32

typedef _Float16 v16h __attribute__((ext_vector_type(16)));
typedef float    v8f  __attribute__((ext_vector_type(8)));

union Frag { v16h v; unsigned u[8]; _Float16 h[16]; };

// ---------------- Kernel 0: zero BN stats ----------------
__global__ void k_zero_stats(float* stats) {
    if (threadIdx.x < 128) stats[threadIdx.x] = 0.0f;
}

// ---------------- Kernel 1: build W1[i,h,v,w] ----------------
// w1 = fc1/||fc1||_col + pos/||pos||_col, pos = rpe[hops]
__global__ void k_w1(const float* __restrict__ fc1, const float* __restrict__ rpe,
                     const int* __restrict__ hops, int rlen, float* __restrict__ w1out) {
    int ih = blockIdx.x;           // 0..23  (i*8+h)
    int w  = threadIdx.x;
    if (w >= VV) return;
    const float* f = fc1 + ih * P2;
    const float* r = rpe + ih * rlen;
    float pv[VV];
    float wn = 0.f, pn = 0.f;
    #pragma unroll
    for (int v = 0; v < VV; ++v) {
        float fv = f[v * VV + w];        wn += fv * fv;
        float p  = r[hops[v * VV + w]];  pv[v] = p;  pn += p * p;
    }
    wn = sqrtf(wn) + 1e-4f;
    pn = sqrtf(pn) + 1e-4f;
    #pragma unroll
    for (int v = 0; v < VV; ++v)
        w1out[ih * P2 + v * VV + w] = f[v * VV + w] / wn + pv[v] / pn;
}

// ---------------- Kernel 2: build D1[n,q,v,w] (q=0..63) ----------------
__global__ void __launch_bounds__(256) k_d1(const float* __restrict__ d,
                                            const float* __restrict__ c1w, const float* __restrict__ c1b,
                                            const float* __restrict__ c4w, const float* __restrict__ c4b,
                                            float* __restrict__ d1out) {
    __shared__ float sc4[HID * HID];
    __shared__ float sc1[HID * 3];
    __shared__ float sb1[HID];
    __shared__ float sb4[HID];
    int tid = threadIdx.x;
    for (int i = tid; i < HID * HID; i += 256) sc4[i] = c4w[i];
    if (tid < HID * 3) sc1[tid] = c1w[tid];
    if (tid < HID) { sb1[tid] = c1b[tid]; sb4[tid] = c4b[tid]; }
    __syncthreads();

    int n = blockIdx.x / 3, part = blockIdx.x % 3;
    int p = part * 256 + tid;
    if (p >= P2) return;
    float d0 = d[((size_t)n * 3 + 0) * P2 + p];
    float d1v = d[((size_t)n * 3 + 1) * P2 + p];
    float d2 = d[((size_t)n * 3 + 2) * P2 + p];
    float th[HID];
    #pragma unroll
    for (int h = 0; h < HID; ++h)
        th[h] = tanhf(sc1[h * 3] * d0 + sc1[h * 3 + 1] * d1v + sc1[h * 3 + 2] * d2 + sb1[h]);
    for (int q = 0; q < HID; ++q) {
        float acc = sb4[q];
        #pragma unroll
        for (int h = 0; h < HID; ++h) acc += th[h] * sc4[q * HID + h];
        d1out[((size_t)n * HID + q) * P2 + p] = acc;
    }
}

// ---------------- Kernel 3: main WMMA kernel, one block per (n,h) ----------------
// y[n, h*8+o, t, w] = sum_c X_c(t,:) . M_{o,c}(:,w) + bias_o
// M_{o,c} = sS[o,c]*D1_c + sum_i sF[i,o,c]*W1_i
__global__ void __launch_bounds__(256) k_main(const float* __restrict__ x,
                                              const float* __restrict__ d1,
                                              const float* __restrict__ w1,
                                              const float* __restrict__ fc2w,
                                              const float* __restrict__ fc2b,
                                              float* __restrict__ y,
                                              float* __restrict__ stats) {
    extern __shared__ char smem[];
    _Float16* sX  = (_Float16*)smem;                 // [TT][KPAD]   128*256 f16 = 64 KB
    _Float16* sB  = sX + TT * KPAD;                  // [VPAD][KPAD] 32*256 f16 = 16 KB (col-major: [w][k])
    float* sD1    = (float*)(sB + VPAD * KPAD);      // 8*625
    float* sW1    = sD1 + 8 * P2;                    // 3*625
    float* sF     = sW1 + 3 * P2;                    // 3*8*8
    float* sS     = sF + 192;                        // 64
    float* sBias  = sS + 64;                         // 8

    const int n   = blockIdx.x >> 3;
    const int h   = blockIdx.x & 7;
    const int tid = threadIdx.x;
    const int wv  = tid >> 5;        // wave id 0..7  -> T-tile
    const int lane = tid & 31;
    const int m   = lane & 15;
    const int hi  = lane >> 4;

    // ---- stage X as f16, zero-padded to KPAD ----
    for (int idx = tid; idx < TT * KPAD; idx += 256) {
        int t = idx >> 8, k = idx & 255;
        int c = k >> 5, v = k & 31;
        float val = (v < VV) ? x[((size_t)(n * CC + h * 8 + c) * TT + t) * VV + v] : 0.0f;
        sX[idx] = (_Float16)val;
    }
    // ---- stage D1, W1, fc2 coefs ----
    for (int idx = tid; idx < 8 * P2; idx += 256)
        sD1[idx] = d1[((size_t)n * HID + h * 8 + (idx / P2)) * P2 + (idx % P2)];
    for (int idx = tid; idx < 3 * P2; idx += 256)
        sW1[idx] = w1[((idx / P2) * 8 + h) * P2 + (idx % P2)];
    if (tid < 192) sF[tid] = fc2w[((tid >> 6) * 8 + h) * 64 + (tid & 63)];
    __syncthreads();
    if (tid < 64) sS[tid] = sF[tid] + sF[64 + tid] + sF[128 + tid];
    if (tid < 8)  sBias[tid] = fc2b[h * 8 + tid] + fc2b[64 + h * 8 + tid] + fc2b[128 + h * 8 + tid];
    __syncthreads();

    for (int o = 0; o < 8; ++o) {
        // ---- build M_o in LDS (f16, [w][k] layout) ----
        {
            int k = tid;                 // 0..255
            int c = k >> 5, v = k & 31;
            float s  = sS[o * 8 + c];
            float f0 = sF[o * 8 + c], f1 = sF[64 + o * 8 + c], f2 = sF[128 + o * 8 + c];
            for (int w = 0; w < VPAD; ++w) {
                float val = 0.0f;
                if (v < VV && w < VV) {
                    int p = v * VV + w;
                    val = s * sD1[c * P2 + p] + f0 * sW1[p] + f1 * sW1[P2 + p] + f2 * sW1[2 * P2 + p];
                }
                sB[w * KPAD + k] = (_Float16)val;
            }
        }
        __syncthreads();

        // ---- WMMA: wave wv owns T-tile [wv*16, wv*16+16), 2 N-tiles, 8 K-steps ----
        v8f acc0 = {0.f,0.f,0.f,0.f,0.f,0.f,0.f,0.f};
        v8f acc1 = {0.f,0.f,0.f,0.f,0.f,0.f,0.f,0.f};
        const int row0 = wv * 16;
        const _Float16* xrow = sX + (row0 + m) * KPAD;
        const _Float16* bcol0 = sB + m * KPAD;
        const _Float16* bcol1 = sB + (16 + m) * KPAD;
        for (int kc = 0; kc < 8; ++kc) {
            int kb = kc * 32;
            Frag a, b0, b1;
            #pragma unroll
            for (int r = 0; r < 8; ++r) {
                int k0 = kb + 2 * r + ((r >= 4) ? 8 : 0) + hi * 8;
                a.u[r]  = *reinterpret_cast<const unsigned*>(xrow  + k0);
                b0.u[r] = *reinterpret_cast<const unsigned*>(bcol0 + k0);
                b1.u[r] = *reinterpret_cast<const unsigned*>(bcol1 + k0);
            }
            acc0 = __builtin_amdgcn_wmma_f32_16x16x32_f16(false, a.v, false, b0.v, (short)0, acc0, false, false);
            acc1 = __builtin_amdgcn_wmma_f32_16x16x32_f16(false, a.v, false, b1.v, (short)0, acc1, false, false);
        }

        // ---- write y + per-channel stats ----
        const float bias = sBias[o];
        const int ch = h * 8 + o;
        float s1 = 0.f, s2 = 0.f;
        float* yrow = y + (size_t)(n * CC + ch) * TT * VV;
        #pragma unroll
        for (int r = 0; r < 8; ++r) {
            int t = row0 + r + hi * 8;
            float v0 = acc0[r] + bias;
            yrow[t * VV + m] = v0;  s1 += v0;  s2 += v0 * v0;
            int col1 = 16 + m;
            if (col1 < VV) {
                float v1 = acc1[r] + bias;
                yrow[t * VV + col1] = v1;  s1 += v1;  s2 += v1 * v1;
            }
        }
        #pragma unroll
        for (int off = 16; off >= 1; off >>= 1) {
            s1 += __shfl_xor(s1, off, 32);
            s2 += __shfl_xor(s2, off, 32);
        }
        if (lane == 0) {
            atomicAdd(&stats[2 * ch], s1);
            atomicAdd(&stats[2 * ch + 1], s2);
        }
        __syncthreads();
    }
}

// ---------------- Kernel 4: BN + residual + ReLU ----------------
__global__ void __launch_bounds__(256) k_final(const float* __restrict__ y,
                                               const float* __restrict__ x,
                                               const float* __restrict__ stats,
                                               const float* __restrict__ g,
                                               const float* __restrict__ b,
                                               float* __restrict__ out) {
    size_t idx = (size_t)blockIdx.x * 256 + threadIdx.x;
    const size_t total = (size_t)NB * CC * TT * VV;
    if (idx >= total) return;
    int ch = (int)((idx / (TT * VV)) % CC);
    const float cnt = (float)NB * TT * VV;     // 409600
    float mean = stats[2 * ch] / cnt;
    float var  = stats[2 * ch + 1] / cnt - mean * mean;
    float inv  = rsqrtf(var + 1e-5f);
    float v = (y[idx] - mean) * inv * g[ch] + b[ch] + x[idx];
    out[idx] = v > 0.f ? v : 0.f;
}

// ---------------- launcher ----------------
extern "C" void kernel_launch(void* const* d_in, const int* in_sizes, int n_in,
                              void* d_out, int out_size, void* d_ws, size_t ws_size,
                              hipStream_t stream) {
    const float* x      = (const float*)d_in[0];
    const float* d      = (const float*)d_in[1];
    // d_in[2] = s (int scalar, ==1, unused)
    const float* fc1    = (const float*)d_in[3];
    const float* rpe    = (const float*)d_in[4];
    const float* c1w    = (const float*)d_in[5];
    const float* c1b    = (const float*)d_in[6];
    const float* c4w    = (const float*)d_in[7];
    const float* c4b    = (const float*)d_in[8];
    const float* fc2w   = (const float*)d_in[9];
    const float* fc2b   = (const float*)d_in[10];
    const float* bng    = (const float*)d_in[11];
    const float* bnb    = (const float*)d_in[12];
    const int*   hops   = (const int*)d_in[13];
    float* out = (float*)d_out;

    const int rlen = in_sizes[4] / 24;          // rpe last-dim (max_hop+1)

    // workspace carve (floats)
    float* ws = (float*)d_ws;
    float* w1_ws    = ws;                                   // 3*8*625  = 15000
    float* d1_ws    = ws + 15360;                           // 128*64*625 = 5,120,000
    float* y_ws     = d1_ws + (size_t)NB * HID * P2;        // 128*64*128*25 = 26,214,400
    float* stats_ws = y_ws + (size_t)NB * CC * TT * VV;     // 128

    k_zero_stats<<<1, 128, 0, stream>>>(stats_ws);
    k_w1<<<24, 32, 0, stream>>>(fc1, rpe, hops, rlen, w1_ws);
    k_d1<<<NB * 3, 256, 0, stream>>>(d, c1w, c1b, c4w, c4b, d1_ws);

    const size_t shmem = (size_t)TT * KPAD * 2 + (size_t)VPAD * KPAD * 2
                       + (8 * P2 + 3 * P2 + 192 + 64 + 8) * sizeof(float);
    k_main<<<NB * HH, 256, shmem, stream>>>(x, d1_ws, w1_ws, fc2w, fc2b, y_ws, stats_ws);

    const size_t total = (size_t)NB * CC * TT * VV;
    k_final<<<(unsigned)((total + 255) / 256), 256, 0, stream>>>(y_ws, x, stats_ws, bng, bnb, out);
}
